// SlidingWindowAttention_9457517985812
// MI455X (gfx1250) — compile-verified, matrix operand order checked by reading, not run on hardware
//
#include <hip/hip_runtime.h>
#include <hip/hip_bf16.h>

typedef __bf16 bf16_t;
typedef __attribute__((ext_vector_type(16))) __bf16 v16bf;
typedef __attribute__((ext_vector_type(8)))  __bf16 v8bf;
typedef __attribute__((ext_vector_type(8)))  float  v8f;

#define WMMA_BF16(a, b, c) \
  __builtin_amdgcn_wmma_f32_16x16x32_bf16(false, (a), false, (b), (short)0, (c), false, false)

#define SWA_WINDOW 512

// ---------------------------------------------------------------------------
// 16-bit A/B fragment loader (CDNA5 ISA 7.12.2, 16-bit 16x32 layout).
// Works for global or LDS (generic) pointers; two 16B loads per lane.
// ---------------------------------------------------------------------------
__device__ __forceinline__ v16bf load_frag(const bf16_t* __restrict__ base,
                                           int row, int ld, int k0) {
  int lane = threadIdx.x & 31;
  int koff = (lane < 16) ? 0 : 8;
  const bf16_t* p = base + (size_t)row * ld + k0 + koff;
  v8bf lo = *(const v8bf*)(p);
  v8bf hi = *(const v8bf*)(p + 16);
  v16bf f;
#pragma unroll
  for (int i = 0; i < 8; ++i) { f[i] = lo[i]; f[8 + i] = hi[i]; }
  return f;
}

// Fragment set for a 32M x 64N wave tile at one K-chunk.
struct Frags { v16bf a[2]; v16bf b[4]; };

__device__ __forceinline__ void load_frags(Frags& f, const bf16_t* __restrict__ A,
                                           const bf16_t* __restrict__ BT,
                                           int ar0, int ar1, int n0, int cc,
                                           int K, int k0) {
  f.a[0] = load_frag(A, ar0, K, k0);
  f.a[1] = load_frag(A, ar1, K, k0);
#pragma unroll
  for (int t = 0; t < 4; ++t) f.b[t] = load_frag(BT, n0 + t * 16 + cc, K, k0);
}

__device__ __forceinline__ void do_wmma(v8f acc[2][4], const Frags& f) {
#pragma unroll
  for (int mt = 0; mt < 2; ++mt)
#pragma unroll
    for (int t = 0; t < 4; ++t)
      acc[mt][t] = WMMA_BF16(f.a[mt], f.b[t], acc[mt][t]);
}

// ---------------------------------------------------------------------------
// Converters
// ---------------------------------------------------------------------------
__global__ void cvt_bf16_kernel(const float* __restrict__ in, bf16_t* __restrict__ out, int n) {
  int i = blockIdx.x * blockDim.x + threadIdx.x;
  if (i < n) out[i] = (bf16_t)in[i];
}

// out[n*K + k] = (bf16) in[k*N + n]   (B^T for contiguous B-fragment rows)
__global__ void transpose_cvt_kernel(const float* __restrict__ in, bf16_t* __restrict__ out,
                                     int K, int N) {
  int i = blockIdx.x * blockDim.x + threadIdx.x;
  if (i < K * N) {
    int n = i / K;
    int k = i - n * K;
    out[i] = (bf16_t)in[(size_t)k * N + n];
  }
}

// ---------------------------------------------------------------------------
// QKV projection: x(4096x1024) @ w_qkv(1024x3072). Wave tile 32M x 64N,
// ping-pong register double buffering (no rotate movs). Scatter epilogue:
// Q,K row-major (bh,s,64), V transposed (bh,64,s). grid(128,12) block 128.
// ---------------------------------------------------------------------------
__global__ __launch_bounds__(128) void qkv_gemm_kernel(
    const bf16_t* __restrict__ A, const bf16_t* __restrict__ BT,
    bf16_t* __restrict__ Qb, bf16_t* __restrict__ Kb, bf16_t* __restrict__ Vt) {
  const int K = 1024;
  int lane = threadIdx.x & 31;
  int wv = threadIdx.x >> 5;
  int m0 = blockIdx.x * 32;
  int n0 = blockIdx.y * 256 + wv * 64;
  int cc = lane & 15;
  int hg = lane >> 4;

  v8f acc[2][4] = {};
  int ar0 = m0 + cc, ar1 = m0 + 16 + cc;

  Frags f0, f1;
  load_frags(f0, A, BT, ar0, ar1, n0, cc, K, 0);
  for (int k0 = 0; k0 < K; k0 += 64) {
    load_frags(f1, A, BT, ar0, ar1, n0, cc, K, k0 + 32);  // k0+32 <= K-32 always
    do_wmma(acc, f0);
    if (k0 + 64 < K) load_frags(f0, A, BT, ar0, ar1, n0, cc, K, k0 + 64);
    do_wmma(acc, f1);
  }

#pragma unroll
  for (int mt = 0; mt < 2; ++mt) {
#pragma unroll
    for (int t = 0; t < 4; ++t) {
      int n = n0 + t * 16 + cc;
      int which = n >> 10;          // 0=q 1=k 2=v
      int h = (n >> 6) & 15;
      int d = n & 63;
#pragma unroll
      for (int r = 0; r < 8; ++r) {
        int m = m0 + mt * 16 + r + 8 * hg;
        int b = m >> 11;
        int s = m & 2047;
        int bh = (b << 4) | h;
        bf16_t val = (bf16_t)acc[mt][t][r];
        if (which == 0)      Qb[((size_t)bh * 2048 + s) * 64 + d] = val;
        else if (which == 1) Kb[((size_t)bh * 2048 + s) * 64 + d] = val;
        else                 Vt[((size_t)bh * 64 + d) * 2048 + s] = val;
      }
    }
  }
}

// ---------------------------------------------------------------------------
// RoPE on (bh=32, s=2048, d=64) bf16, interleaved even/odd pairs.
// ---------------------------------------------------------------------------
__global__ void rope_kernel(bf16_t* __restrict__ qk) {
  int idx = blockIdx.x * blockDim.x + threadIdx.x;
  int i  = idx & 31;
  int s  = (idx >> 5) & 2047;
  int bh = idx >> 16;
  float inv = __powf(10000.0f, -(float)i * (1.0f / 32.0f));
  float ang = (float)s * inv;
  float sn, c;
  __sincosf(ang, &sn, &c);
  bf16_t* p = qk + ((size_t)bh * 2048 + s) * 64 + 2 * i;
  float e = (float)p[0], o = (float)p[1];
  p[0] = (bf16_t)(e * c - o * sn);
  p[1] = (bf16_t)(o * c + e * sn);
}

// ---------------------------------------------------------------------------
// Async global->LDS staging of one 32-key K tile (32x64 bf16, contiguous
// 4 KB) and one V tile (64 rows x 32 cols bf16, 64B per row) using the
// CDNA5 async-copy engine (ASYNCcnt). Wave-private buffers, no barriers.
// ---------------------------------------------------------------------------
__device__ __forceinline__ void stage_async(const bf16_t* __restrict__ kg,
                                            const bf16_t* __restrict__ vg,
                                            __bf16* kl, __bf16* vl, int lane) {
  const char* kgc = (const char*)kg;
  const char* vgc = (const char*)vg;
  uint32_t klo = (uint32_t)(uintptr_t)kl;   // low 32 bits of generic ptr = LDS addr
  uint32_t vlo = (uint32_t)(uintptr_t)vl;
#pragma unroll
  for (int i = 0; i < 8; ++i) {             // K: 8 x (32 lanes x 16B) = 4 KB
    uint64_t ga = (uint64_t)(uintptr_t)(kgc + i * 512 + lane * 16);
    uint32_t la = klo + i * 512 + lane * 16;
    asm volatile("global_load_async_to_lds_b128 %0, %1, off"
                 :: "v"(la), "v"(ga) : "memory");
  }
#pragma unroll
  for (int i = 0; i < 8; ++i) {             // V: 256 x 16B chunks, 4 per row
    int j = i * 32 + lane;
    int row = j >> 2;
    int c16 = j & 3;
    uint64_t ga = (uint64_t)(uintptr_t)(vgc + (size_t)row * 4096 + c16 * 16);
    uint32_t la = vlo + j * 16;
    asm volatile("global_load_async_to_lds_b128 %0, %1, off"
                 :: "v"(la), "v"(ga) : "memory");
  }
}

// ---------------------------------------------------------------------------
// Sliding-window flash attention. One wave per (bh, 16-query tile).
// K/V tiles double-buffered in LDS via async copies; per 32-key chunk:
// 4x WMMA (QK^T) -> online softmax -> P via LDS -> 4x WMMA (PV).
// ---------------------------------------------------------------------------
__global__ __launch_bounds__(128) void attn_kernel(
    const bf16_t* __restrict__ Q, const bf16_t* __restrict__ Kk,
    const bf16_t* __restrict__ Vt, bf16_t* __restrict__ O) {
  // per wave, per parity: K tile (32x64) then V tile (64x32)
  __shared__ __align__(16) __bf16 kvbuf[4][2][32 * 64 + 64 * 32];
  __shared__ __align__(16) __bf16 pbuf[4][16][32];

  int lane = threadIdx.x & 31;
  int wv = threadIdx.x >> 5;
  int wid = blockIdx.x * 4 + wv;
  int qt = wid & 127;
  int bh = wid >> 7;
  int q0 = qt << 4;
  int cc = lane & 15;
  int hg = lane >> 4;
  const float scale = 0.125f;

  const bf16_t* Qp = Q  + (size_t)bh * (2048 * 64);
  const bf16_t* Kp = Kk + (size_t)bh * (2048 * 64);
  const bf16_t* Vp = Vt + (size_t)bh * (64 * 2048);

  v16bf qa0 = load_frag(Qp, q0 + cc, 64, 0);
  v16bf qa1 = load_frag(Qp, q0 + cc, 64, 32);

  v8f o0 = {}, o1 = {}, o2 = {}, o3 = {};
  float mrow[8], lrow[8];
#pragma unroll
  for (int r = 0; r < 8; ++r) { mrow[r] = -1e30f; lrow[r] = 0.0f; }

  int kc0 = q0 - SWA_WINDOW;
  if (kc0 < 0) kc0 = 0;
  kc0 &= ~31;

  int par = 0;
  stage_async(Kp + (size_t)kc0 * 64, Vp + kc0,
              &kvbuf[wv][0][0], &kvbuf[wv][0][32 * 64], lane);

  for (int kc = kc0; kc <= q0; kc += 32) {
    bool more = (kc + 32 <= q0);
    if (more)
      stage_async(Kp + (size_t)(kc + 32) * 64, Vp + (kc + 32),
                  &kvbuf[wv][par ^ 1][0], &kvbuf[wv][par ^ 1][32 * 64], lane);
    // Wait for current tile (in-order ASYNCcnt; next tile's 16 ops may remain)
    if (more) asm volatile("s_wait_asynccnt 16" ::: "memory");
    else      asm volatile("s_wait_asynccnt 0"  ::: "memory");

    const __bf16* kl = &kvbuf[wv][par][0];
    const __bf16* vl = &kvbuf[wv][par][32 * 64];

    v16bf kb00 = load_frag(kl, cc,      64, 0);
    v16bf kb01 = load_frag(kl, cc,      64, 32);
    v16bf kb10 = load_frag(kl, 16 + cc, 64, 0);
    v16bf kb11 = load_frag(kl, 16 + cc, 64, 32);
    v8f s0 = {}; s0 = WMMA_BF16(qa0, kb00, s0); s0 = WMMA_BF16(qa1, kb01, s0);
    v8f s1 = {}; s1 = WMMA_BF16(qa0, kb10, s1); s1 = WMMA_BF16(qa1, kb11, s1);

    float al[8];
#pragma unroll
    for (int r = 0; r < 8; ++r) {
      int row = q0 + r + 8 * hg;
      int c0 = kc + cc;
      int c1 = c0 + 16;
      bool ok0 = (c0 <= row) && (c0 + SWA_WINDOW >= row);
      bool ok1 = (c1 <= row) && (c1 + SWA_WINDOW >= row);
      float sv0 = ok0 ? s0[r] * scale : -1e30f;
      float sv1 = ok1 ? s1[r] * scale : -1e30f;
      float mx = fmaxf(sv0, sv1);
#pragma unroll
      for (int d = 1; d < 16; d <<= 1) mx = fmaxf(mx, __shfl_xor(mx, d, 32));
      float mnew = fmaxf(mrow[r], mx);
      float a = __expf(mrow[r] - mnew);
      mrow[r] = mnew;
      float e0 = ok0 ? __expf(sv0 - mnew) : 0.0f;
      float e1 = ok1 ? __expf(sv1 - mnew) : 0.0f;
      float rs = e0 + e1;
#pragma unroll
      for (int d = 1; d < 16; d <<= 1) rs += __shfl_xor(rs, d, 32);
      lrow[r] = lrow[r] * a + rs;
      al[r] = a;
      int rowr = r + 8 * hg;
      pbuf[wv][rowr][cc]      = (__bf16)e0;
      pbuf[wv][rowr][16 + cc] = (__bf16)e1;
    }
#pragma unroll
    for (int r = 0; r < 8; ++r) {
      o0[r] *= al[r]; o1[r] *= al[r]; o2[r] *= al[r]; o3[r] *= al[r];
    }
    v16bf pf  = load_frag((const bf16_t*)&pbuf[wv][0][0], cc, 32, 0);
    v16bf vb0 = load_frag(vl,  0 + cc, 32, 0);
    v16bf vb1 = load_frag(vl, 16 + cc, 32, 0);
    v16bf vb2 = load_frag(vl, 32 + cc, 32, 0);
    v16bf vb3 = load_frag(vl, 48 + cc, 32, 0);
    o0 = WMMA_BF16(pf, vb0, o0);
    o1 = WMMA_BF16(pf, vb1, o1);
    o2 = WMMA_BF16(pf, vb2, o2);
    o3 = WMMA_BF16(pf, vb3, o3);
    par ^= 1;
  }

  int b = bh >> 4, h = bh & 15;
#pragma unroll
  for (int r = 0; r < 8; ++r) {
    int s = q0 + r + 8 * hg;
    size_t rowoff = ((size_t)b * 2048 + s) * 1024 + h * 64;
    float inv_l = 1.0f / lrow[r];
    O[rowoff +  0 + cc] = (bf16_t)(o0[r] * inv_l);
    O[rowoff + 16 + cc] = (bf16_t)(o1[r] * inv_l);
    O[rowoff + 32 + cc] = (bf16_t)(o2[r] * inv_l);
    O[rowoff + 48 + cc] = (bf16_t)(o3[r] * inv_l);
  }
}

// ---------------------------------------------------------------------------
// Output projection: attn(4096x1024) @ w_o(1024x1024) -> f32. Wave tile
// 32x64, ping-pong register double buffering. grid(128,4) block 128.
// ---------------------------------------------------------------------------
__global__ __launch_bounds__(128) void out_gemm_kernel(
    const bf16_t* __restrict__ A, const bf16_t* __restrict__ BT, float* __restrict__ C) {
  const int K = 1024, N = 1024;
  int lane = threadIdx.x & 31;
  int wv = threadIdx.x >> 5;
  int m0 = blockIdx.x * 32;
  int n0 = blockIdx.y * 256 + wv * 64;
  int cc = lane & 15;
  int hg = lane >> 4;

  v8f acc[2][4] = {};
  int ar0 = m0 + cc, ar1 = m0 + 16 + cc;

  Frags f0, f1;
  load_frags(f0, A, BT, ar0, ar1, n0, cc, K, 0);
  for (int k0 = 0; k0 < K; k0 += 64) {
    load_frags(f1, A, BT, ar0, ar1, n0, cc, K, k0 + 32);
    do_wmma(acc, f0);
    if (k0 + 64 < K) load_frags(f0, A, BT, ar0, ar1, n0, cc, K, k0 + 64);
    do_wmma(acc, f1);
  }

#pragma unroll
  for (int mt = 0; mt < 2; ++mt)
#pragma unroll
    for (int t = 0; t < 4; ++t)
#pragma unroll
      for (int r = 0; r < 8; ++r) {
        int m = m0 + mt * 16 + r + 8 * hg;
        C[(size_t)m * N + n0 + t * 16 + cc] = acc[mt][t][r];
      }
}

// ---------------------------------------------------------------------------
extern "C" void kernel_launch(void* const* d_in, const int* in_sizes, int n_in,
                              void* d_out, int out_size, void* d_ws, size_t ws_size,
                              hipStream_t stream) {
  const float* x     = (const float*)d_in[0];   // (2,2048,1024)
  const float* w_qkv = (const float*)d_in[1];   // (1024,3072)
  const float* w_o   = (const float*)d_in[2];   // (1024,1024)
  float* out = (float*)d_out;                   // (2,2048,1024)

  bf16_t* xb    = (bf16_t*)d_ws;                 // 4096*1024
  bf16_t* wqkvT = xb    + 4096 * 1024;           // 3072*1024  (w_qkv^T)
  bf16_t* woT   = wqkvT + 3072 * 1024;           // 1024*1024  (w_o^T)
  bf16_t* Qb    = woT   + 1024 * 1024;           // 32*2048*64
  bf16_t* Kb    = Qb    + 32 * 2048 * 64;        // 32*2048*64
  bf16_t* Vt    = Kb    + 32 * 2048 * 64;        // 32*64*2048 (V transposed)
  bf16_t* attn  = Vt    + 32 * 2048 * 64;        // 4096*1024

  cvt_bf16_kernel<<<(4096 * 1024 + 255) / 256, 256, 0, stream>>>(x, xb, 4096 * 1024);
  transpose_cvt_kernel<<<(3072 * 1024 + 255) / 256, 256, 0, stream>>>(w_qkv, wqkvT, 1024, 3072);
  transpose_cvt_kernel<<<(1024 * 1024 + 255) / 256, 256, 0, stream>>>(w_o, woT, 1024, 1024);

  qkv_gemm_kernel<<<dim3(128, 12), 128, 0, stream>>>(xb, wqkvT, Qb, Kb, Vt);

  rope_kernel<<<8192, 256, 0, stream>>>(Qb);
  rope_kernel<<<8192, 256, 0, stream>>>(Kb);

  attn_kernel<<<1024, 128, 0, stream>>>(Qb, Kb, Vt, attn);

  out_gemm_kernel<<<dim3(128, 4), 128, 0, stream>>>(attn, woT, out);
}
